// Attention_90924457656780
// MI455X (gfx1250) — matrix operand order repaired and checked by reading, not compiled, not run on hardware
//
#include <hip/hip_runtime.h>

// ---- problem constants ----
#define Bb   2
#define Ss   2048
#define Dd   4096
#define Hh   32
#define HKVv 8
#define HDd  128
#define Mrows (Bb*Ss)   // 4096

typedef __attribute__((ext_vector_type(16))) __bf16 v16bf;
typedef __attribute__((ext_vector_type(8)))  float  v8f;

union Frag16 {
    uint4 q[2];
    v16bf v;
    unsigned short u[16];
};

__device__ __forceinline__ unsigned short f2bf(float f) {
    unsigned u = __float_as_uint(f);
    u += 0x7FFFu + ((u >> 16) & 1u);   // round-to-nearest-even
    return (unsigned short)(u >> 16);
}

__device__ __forceinline__ float redmax16(float v) {
    v = fmaxf(v, __shfl_xor(v, 1, 32));
    v = fmaxf(v, __shfl_xor(v, 2, 32));
    v = fmaxf(v, __shfl_xor(v, 4, 32));
    v = fmaxf(v, __shfl_xor(v, 8, 32));
    return v;
}
__device__ __forceinline__ float redsum16(float v) {
    v += __shfl_xor(v, 1, 32);
    v += __shfl_xor(v, 2, 32);
    v += __shfl_xor(v, 4, 32);
    v += __shfl_xor(v, 8, 32);
    return v;
}

// CDNA5 async global->LDS copy (16B per lane), tracked by ASYNCcnt.
__device__ __forceinline__ void async_copy16(void* lds_dst, const void* gsrc) {
    unsigned loff = (unsigned)(uintptr_t)lds_dst;   // low 32 bits = LDS offset
    asm volatile("global_load_async_to_lds_b128 %0, %1, off"
                 :: "v"(loff), "v"(gsrc)
                 : "memory");
}
__device__ __forceinline__ void wait_async0() {
    asm volatile("s_wait_asynccnt 0" ::: "memory");
}

// ---------------------------------------------------------------------------
// fp32 -> bf16 conversion
// ---------------------------------------------------------------------------
__global__ void cvt_f32_bf16(const float* __restrict__ s,
                             unsigned short* __restrict__ d, int n) {
    for (int i = blockIdx.x * blockDim.x + threadIdx.x; i < n;
         i += gridDim.x * blockDim.x)
        d[i] = f2bf(s[i]);
}

// ---------------------------------------------------------------------------
// C[M,N] = (A[M,K] * B[N,K]^T + bias) * outScale   (A,B bf16; C bf16 or f32)
// block tile 128(M) x 128(N), K-step 32; 8 waves = 2(M) x 4(N), each wave a
// 64x32 subtile -> 8 WMMAs per K-step.  Double-buffered async global->LDS
// staging; one barrier per K-step.
// ---------------------------------------------------------------------------
template <bool OUT_BF16>
__global__ __launch_bounds__(256)
void gemm_xwT(const unsigned short* __restrict__ A,
              const unsigned short* __restrict__ Bw,
              const float* __restrict__ bias,
              void* __restrict__ Cout, int M, int N, int K, float outScale) {
    __shared__ __align__(16) unsigned short As[2][128 * 32];
    __shared__ __align__(16) unsigned short Bs[2][128 * 32];

    const int tid  = threadIdx.x;
    const int lane = tid & 31;
    const int w    = tid >> 5;
    const int l15  = lane & 15;
    const int hl   = (lane >> 4) & 1;
    const int wmi  = w >> 2;       // 0..1 along M (64 rows each)
    const int wni  = w & 3;        // 0..3 along N (32 cols each)
    const int m0   = blockIdx.y * 128;
    const int n0   = blockIdx.x * 128;

    const int rt = tid >> 2, ct = tid & 3;     // staging chunk coords

    v8f C[4][2] = {};

    // prologue: stage tile k0=0 into buffer 0
    async_copy16(&As[0][rt * 32 + ct * 8],
                 &A[(size_t)(m0 + rt) * K + ct * 8]);
    async_copy16(&As[0][(rt + 64) * 32 + ct * 8],
                 &A[(size_t)(m0 + rt + 64) * K + ct * 8]);
    async_copy16(&Bs[0][rt * 32 + ct * 8],
                 &Bw[(size_t)(n0 + rt) * K + ct * 8]);
    async_copy16(&Bs[0][(rt + 64) * 32 + ct * 8],
                 &Bw[(size_t)(n0 + rt + 64) * K + ct * 8]);

    const int nsteps = K / 32;
    for (int t = 0; t < nsteps; ++t) {
        const int buf = t & 1;
        wait_async0();
        __syncthreads();           // tile t visible to all waves

        if (t + 1 < nsteps) {      // stage tile t+1 into other buffer
            const int k1 = (t + 1) * 32;
            async_copy16(&As[buf ^ 1][rt * 32 + ct * 8],
                         &A[(size_t)(m0 + rt) * K + k1 + ct * 8]);
            async_copy16(&As[buf ^ 1][(rt + 64) * 32 + ct * 8],
                         &A[(size_t)(m0 + rt + 64) * K + k1 + ct * 8]);
            async_copy16(&Bs[buf ^ 1][rt * 32 + ct * 8],
                         &Bw[(size_t)(n0 + rt) * K + k1 + ct * 8]);
            async_copy16(&Bs[buf ^ 1][(rt + 64) * 32 + ct * 8],
                         &Bw[(size_t)(n0 + rt + 64) * K + k1 + ct * 8]);
        }

        const unsigned short* Asb = As[buf];
        const unsigned short* Bsb = Bs[buf];
        Frag16 a[4], b[2];
#pragma unroll
        for (int ai = 0; ai < 4; ai++) {
            int row = wmi * 64 + ai * 16 + l15;          // lane = M
            a[ai].q[0] = *(const uint4*)&Asb[row * 32 + hl * 8];
            a[ai].q[1] = *(const uint4*)&Asb[row * 32 + hl * 8 + 16];
        }
#pragma unroll
        for (int bj = 0; bj < 2; bj++) {
            int row = wni * 32 + bj * 16 + l15;          // lane = N
            b[bj].q[0] = *(const uint4*)&Bsb[row * 32 + hl * 8];
            b[bj].q[1] = *(const uint4*)&Bsb[row * 32 + hl * 8 + 16];
        }
#pragma unroll
        for (int ai = 0; ai < 4; ai++)
#pragma unroll
            for (int bj = 0; bj < 2; bj++)
                C[ai][bj] = __builtin_amdgcn_wmma_f32_16x16x32_bf16(
                    false, a[ai].v, false, b[bj].v, (short)0, C[ai][bj],
                    false, false);
        // no trailing barrier: next iteration's wait+barrier orders buffer reuse
    }

#pragma unroll
    for (int ai = 0; ai < 4; ai++)
#pragma unroll
        for (int bj = 0; bj < 2; bj++) {
            int n = n0 + wni * 32 + bj * 16 + l15;
            float bval = bias ? bias[n] : 0.0f;
#pragma unroll
            for (int i = 0; i < 8; i++) {
                int m = m0 + wmi * 64 + ai * 16 + i + hl * 8;
                float val = (C[ai][bj][i] + bval) * outScale;
                if (OUT_BF16)
                    ((unsigned short*)Cout)[(size_t)m * N + n] = f2bf(val);
                else
                    ((float*)Cout)[(size_t)m * N + n] = val;
            }
        }
}

// ---------------------------------------------------------------------------
// causal flash attention; grid = (S/128, H, B), 256 threads (8 waves)
// wave w owns Q rows [q0+16w, q0+16w+16); KV tiles of 32 keys.
// K tile: double-buffered async global->LDS.  V tile: double-buffered via
// VGPR transpose (global loads for t+1 issued right after the barrier).
// Q is pre-scaled by 1/sqrt(HD) in its projection epilogue.
// ---------------------------------------------------------------------------
__global__ __launch_bounds__(256)
void flash_attn(const unsigned short* __restrict__ Q,
                const unsigned short* __restrict__ Kg,
                const unsigned short* __restrict__ Vg,
                unsigned short* __restrict__ O) {
    __shared__ __align__(16) unsigned short Kt[2][32 * 128];   // [key][hd]
    __shared__ __align__(16) unsigned short Vt[2][128 * 32];   // [hd][key]
    __shared__ __align__(16) unsigned short Pb[8 * 16 * 32];   // per-wave P

    const int tid  = threadIdx.x;
    const int lane = tid & 31;
    const int w    = tid >> 5;
    const int l15  = lane & 15;
    const int hl   = (lane >> 4) & 1;
    const int b    = blockIdx.z;
    const int h    = blockIdx.y;
    const int q0   = blockIdx.x * 128;
    const int kvh  = h >> 2;                 // GQA: rep = H/HKV = 4

    // Q A-fragments: 4 K-chunks of 32 over HD=128
    Frag16 qf[4];
    {
        const unsigned short* qrow =
            Q + (size_t)(b * Ss + q0 + w * 16 + l15) * (Hh * HDd) + h * HDd;
#pragma unroll
        for (int c = 0; c < 4; c++) {
            qf[c].q[0] = *(const uint4*)&qrow[c * 32 + hl * 8];
            qf[c].q[1] = *(const uint4*)&qrow[c * 32 + hl * 8 + 16];
        }
    }

    v8f Oacc[8] = {};
    float m[8], ssum[8];
#pragma unroll
    for (int i = 0; i < 8; i++) { m[i] = -1e9f; ssum[i] = 0.0f; }

    const unsigned short* Kbase = Kg + (size_t)(b * Ss) * (HKVv * HDd) + kvh * HDd;
    const unsigned short* Vbase = Vg + (size_t)(b * Ss) * (HKVv * HDd) + kvh * HDd;

    const int rr = tid >> 4, cc = tid & 15;  // tile chunk coords (first half)

    // prologue: stage tile 0
    async_copy16(&Kt[0][rr * 128 + cc * 8],
                 &Kbase[(size_t)rr * (HKVv * HDd) + cc * 8]);
    async_copy16(&Kt[0][(rr + 16) * 128 + cc * 8],
                 &Kbase[(size_t)(rr + 16) * (HKVv * HDd) + cc * 8]);
    uint4 vdat[2];
    vdat[0] = *(const uint4*)&Vbase[(size_t)rr * (HKVv * HDd) + cc * 8];
    vdat[1] = *(const uint4*)&Vbase[(size_t)(rr + 16) * (HKVv * HDd) + cc * 8];

    const int ntiles = (q0 + 128) / 32;
    for (int t = 0; t < ntiles; ++t) {
        const int kt  = t * 32;
        const int buf = t & 1;

        // transpose-store V tile t (held in VGPRs) into LDS
#pragma unroll
        for (int half = 0; half < 2; half++) {
            const unsigned short* p = (const unsigned short*)&vdat[half];
            int r = rr + half * 16;
#pragma unroll
            for (int j = 0; j < 8; j++)
                Vt[buf][(cc * 8 + j) * 32 + r] = p[j];
        }

        wait_async0();
        __syncthreads();           // K & V of tile t visible to all waves

        if (t + 1 < ntiles) {      // stage tile t+1 (overlaps compute below)
            const int kn = kt + 32;
            async_copy16(&Kt[buf ^ 1][rr * 128 + cc * 8],
                         &Kbase[(size_t)(kn + rr) * (HKVv * HDd) + cc * 8]);
            async_copy16(&Kt[buf ^ 1][(rr + 16) * 128 + cc * 8],
                         &Kbase[(size_t)(kn + rr + 16) * (HKVv * HDd) + cc * 8]);
            vdat[0] = *(const uint4*)&Vbase[(size_t)(kn + rr) * (HKVv * HDd) + cc * 8];
            vdat[1] = *(const uint4*)&Vbase[(size_t)(kn + rr + 16) * (HKVv * HDd) + cc * 8];
        }

        const unsigned short* Ktb = Kt[buf];
        const unsigned short* Vtb = Vt[buf];

        // scores: S = Q (16x128) x K^T  -> two 16x16 blocks over 32 keys
        v8f S0 = {}, S1 = {};
#pragma unroll
        for (int c = 0; c < 4; c++) {
            Frag16 b0, b1;
            int koff = c * 32 + hl * 8;
            b0.q[0] = *(const uint4*)&Ktb[l15 * 128 + koff];
            b0.q[1] = *(const uint4*)&Ktb[l15 * 128 + koff + 16];
            b1.q[0] = *(const uint4*)&Ktb[(16 + l15) * 128 + koff];
            b1.q[1] = *(const uint4*)&Ktb[(16 + l15) * 128 + koff + 16];
            S0 = __builtin_amdgcn_wmma_f32_16x16x32_bf16(
                false, qf[c].v, false, b0.v, (short)0, S0, false, false);
            S1 = __builtin_amdgcn_wmma_f32_16x16x32_bf16(
                false, qf[c].v, false, b1.v, (short)0, S1, false, false);
        }

        // online softmax; C-layout: row M = i + 8*hl, col N = l15
        const int qg_base = q0 + w * 16 + hl * 8;
        const int k0i = kt + l15;
        const int k1i = kt + 16 + l15;
#pragma unroll
        for (int i = 0; i < 8; i++) {
            int qg = qg_base + i;
            float s0 = (k0i <= qg) ? S0[i] : -1e9f;
            float s1 = (k1i <= qg) ? S1[i] : -1e9f;
            float rm = redmax16(fmaxf(s0, s1));
            float mnew  = fmaxf(m[i], rm);
            float alpha = __expf(m[i] - mnew);
            float p0 = __expf(s0 - mnew);
            float p1 = __expf(s1 - mnew);
            float rs = redsum16(p0 + p1);
            ssum[i] = ssum[i] * alpha + rs;
            m[i] = mnew;
#pragma unroll
            for (int nb = 0; nb < 8; nb++) Oacc[nb][i] *= alpha;
            int prow = i + hl * 8;
            Pb[w * 512 + prow * 32 + l15]      = f2bf(p0);
            Pb[w * 512 + prow * 32 + 16 + l15] = f2bf(p1);
        }

        // P (16x32 keys) as A-fragment; O += P x V
        Frag16 pf;
        pf.q[0] = *(const uint4*)&Pb[w * 512 + l15 * 32 + hl * 8];
        pf.q[1] = *(const uint4*)&Pb[w * 512 + l15 * 32 + hl * 8 + 16];
#pragma unroll
        for (int nb = 0; nb < 8; nb++) {
            Frag16 vf;
            int hd = nb * 16 + l15;
            vf.q[0] = *(const uint4*)&Vtb[hd * 32 + hl * 8];
            vf.q[1] = *(const uint4*)&Vtb[hd * 32 + hl * 8 + 16];
            Oacc[nb] = __builtin_amdgcn_wmma_f32_16x16x32_bf16(
                false, pf.v, false, vf.v, (short)0, Oacc[nb], false, false);
        }
        // no trailing barrier: next iteration's wait+barrier orders reuse
    }

    float inv[8];
#pragma unroll
    for (int i = 0; i < 8; i++) inv[i] = 1.0f / ssum[i];
#pragma unroll
    for (int nb = 0; nb < 8; nb++) {
        int col = h * HDd + nb * 16 + l15;
#pragma unroll
        for (int i = 0; i < 8; i++) {
            int row = b * Ss + q0 + w * 16 + i + hl * 8;
            O[(size_t)row * (Hh * HDd) + col] = f2bf(Oacc[nb][i] * inv[i]);
        }
    }
}

// ---------------------------------------------------------------------------
extern "C" void kernel_launch(void* const* d_in, const int* in_sizes, int n_in,
                              void* d_out, int out_size, void* d_ws, size_t ws_size,
                              hipStream_t stream) {
    (void)in_sizes; (void)n_in; (void)out_size; (void)ws_size;
    const float* x  = (const float*)d_in[0];
    const float* wq = (const float*)d_in[1];
    const float* bq = (const float*)d_in[2];
    const float* wk = (const float*)d_in[3];
    const float* bk = (const float*)d_in[4];
    const float* wv = (const float*)d_in[5];
    const float* bv = (const float*)d_in[6];
    const float* wo = (const float*)d_in[7];
    const float* bo = (const float*)d_in[8];

    char* ws = (char*)d_ws;
    size_t off = 0;
    unsigned short* xb  = (unsigned short*)(ws + off); off += (size_t)Mrows * Dd * 2;          // 32 MB
    unsigned short* wqb = (unsigned short*)(ws + off); off += (size_t)(Hh*HDd) * Dd * 2;       // 32 MB
    unsigned short* wkb = (unsigned short*)(ws + off); off += (size_t)(HKVv*HDd) * Dd * 2;     //  8 MB
    unsigned short* wvb = (unsigned short*)(ws + off); off += (size_t)(HKVv*HDd) * Dd * 2;     //  8 MB
    unsigned short* wob = (unsigned short*)(ws + off); off += (size_t)Dd * (Hh*HDd) * 2;       // 32 MB
    unsigned short* qb  = (unsigned short*)(ws + off); off += (size_t)Mrows * (Hh*HDd) * 2;    // 32 MB
    unsigned short* kb  = (unsigned short*)(ws + off); off += (size_t)Mrows * (HKVv*HDd) * 2;  //  8 MB
    unsigned short* vb  = (unsigned short*)(ws + off); off += (size_t)Mrows * (HKVv*HDd) * 2;  //  8 MB
    unsigned short* ob  = (unsigned short*)(ws + off); off += (size_t)Mrows * (Hh*HDd) * 2;    // 32 MB

    const float qscale = 0.0883883476483184f; // 1/sqrt(128)

    // 1) convert inputs to bf16
    cvt_f32_bf16<<<4096, 256, 0, stream>>>(x,  xb,  Mrows * Dd);
    cvt_f32_bf16<<<4096, 256, 0, stream>>>(wq, wqb, (Hh*HDd) * Dd);
    cvt_f32_bf16<<<1024, 256, 0, stream>>>(wk, wkb, (HKVv*HDd) * Dd);
    cvt_f32_bf16<<<1024, 256, 0, stream>>>(wv, wvb, (HKVv*HDd) * Dd);
    cvt_f32_bf16<<<4096, 256, 0, stream>>>(wo, wob, Dd * (Hh*HDd));

    // 2) Q/K/V projections (Q pre-scaled by 1/sqrt(HD))
    gemm_xwT<true><<<dim3((Hh*HDd)/128, Mrows/128), 256, 0, stream>>>(
        xb, wqb, bq, qb, Mrows, Hh*HDd, Dd, qscale);
    gemm_xwT<true><<<dim3((HKVv*HDd)/128, Mrows/128), 256, 0, stream>>>(
        xb, wkb, bk, kb, Mrows, HKVv*HDd, Dd, 1.0f);
    gemm_xwT<true><<<dim3((HKVv*HDd)/128, Mrows/128), 256, 0, stream>>>(
        xb, wvb, bv, vb, Mrows, HKVv*HDd, Dd, 1.0f);

    // 3) causal flash attention
    flash_attn<<<dim3(Ss/128, Hh, Bb), 256, 0, stream>>>(qb, kb, vb, ob);

    // 4) output projection (fp32 out)
    gemm_xwT<false><<<dim3(Dd/128, Mrows/128), 256, 0, stream>>>(
        ob, wob, bo, d_out, Mrows, Dd, Hh*HDd, 1.0f);
}
